// Seq2Seq_44805098832324
// MI455X (gfx1250) — matrix operand order repaired and checked by reading, not compiled, not run on hardware
//
#include <hip/hip_runtime.h>

// ---------------- problem constants ----------------
static constexpr int kB  = 32;
static constexpr int kS  = 64;
static constexpr int kT  = 48;
static constexpr int kV  = 32000;
static constexpr int kE  = 512;
static constexpr int kH  = 1024;
static constexpr int kH3 = 3 * kH;

typedef __attribute__((ext_vector_type(16))) __bf16 v16bf;
typedef __attribute__((ext_vector_type(8)))  __bf16 v8bf;
typedef __attribute__((ext_vector_type(8)))  float  v8f;

// ---------------- fp32 -> bf16 convert (weights) ----------------
__global__ void cvt_bf16(const float* __restrict__ s, __bf16* __restrict__ d, long n) {
    long i = (long)blockIdx.x * blockDim.x + threadIdx.x;
    long stride = (long)gridDim.x * blockDim.x;
    for (; i < n; i += stride) d[i] = (__bf16)s[i];
}

// ---------------- embedding gather (fp32 table -> bf16 rows) ----------------
__global__ void embed_gather(const int* __restrict__ ids, int id_stride,
                             const float* __restrict__ emb,
                             __bf16* __restrict__ dst, long dst_stride, int ncols) {
    int row = blockIdx.y;
    int col = blockIdx.x * blockDim.x + threadIdx.x;
    if (col >= ncols) return;
    int tok = ids[(long)row * id_stride];
    dst[(long)row * dst_stride + col] = (__bf16)emb[(long)tok * ncols + col];
}

// ---------------- zero recurrent state ----------------
__global__ void zero_state(float* h0, float* h1, __bf16* h0b, __bf16* h1b) {
    int i = blockIdx.x * blockDim.x + threadIdx.x;
    h0[i] = 0.f; h1[i] = 0.f;
    h0b[i] = (__bf16)0.f; h1b[i] = (__bf16)0.f;
}

// ---------------- WMMA bf16 GEMM: C(MxN,f32) = A(MxK) * W(NxK)^T + bias ----------
// One wave per 32x32 output tile: two A fragments (M rows m0..m0+31) share both
// B fragments, so each weight element is loaded exactly ONCE per M=32 strip --
// critical for the weight-bandwidth-bound logits GEMM (B=32 => ~64 FLOP/weight).
// W is stored N x K row-major (transposed) so B-fragment loads are contiguous in K.
// Fragment layouts per CDNA5 ISA 7.12.2 (wave32, v_wmma_f32_16x16x32_bf16).
__global__ __launch_bounds__(32)
void wmma_gemm_bf16(const __bf16* __restrict__ A, long lda,
                    const __bf16* __restrict__ W, long ldw,
                    const float* __restrict__ bias,
                    float* __restrict__ C, long ldc,
                    __bf16* __restrict__ Cbf, long ldcb,
                    int K) {
    const int lane = threadIdx.x;
    const long n0 = (long)blockIdx.x * 32;
    const long m0 = (long)blockIdx.y * 32;

    // A fragment: lane<16 holds row m, K={0..7,16..23}; lane>=16 K={8..15,24..31}
    const int am = lane & 15;
    const int ak = (lane >> 4) * 8;
    // B fragment: lane holds column n=lane&15, K={0..7,16..23} (lane<16) / {8..15,24..31} (lane>=16)
    const int bn = lane & 15;
    const int bk = (lane >> 4) * 8;

    const __bf16* pa0 = A + (m0 + am) * lda + ak;
    const __bf16* pa1 = pa0 + 16 * lda;
    const __bf16* pb0 = W + (n0 +  0 + bn) * ldw + bk;
    const __bf16* pb1 = W + (n0 + 16 + bn) * ldw + bk;

    v8f acc[2][2] = {};

    for (int kb = 0; kb < K; kb += 32) {
        // prefetch streaming weight rows two K-chunks ahead (global_prefetch_b8)
        __builtin_prefetch(pb0 + kb + 64, 0, 1);
        __builtin_prefetch(pb1 + kb + 64, 0, 1);

        union Frag { v16bf v; v8bf h[2]; } a0, a1, b0, b1;
        a0.h[0] = *(const v8bf*)(pa0 + kb);   a0.h[1] = *(const v8bf*)(pa0 + kb + 16);
        a1.h[0] = *(const v8bf*)(pa1 + kb);   a1.h[1] = *(const v8bf*)(pa1 + kb + 16);
        b0.h[0] = *(const v8bf*)(pb0 + kb);   b0.h[1] = *(const v8bf*)(pb0 + kb + 16);
        b1.h[0] = *(const v8bf*)(pb1 + kb);   b1.h[1] = *(const v8bf*)(pb1 + kb + 16);

        acc[0][0] = __builtin_amdgcn_wmma_f32_16x16x32_bf16(false, a0.v, false, b0.v, (short)0, acc[0][0], false, false);
        acc[0][1] = __builtin_amdgcn_wmma_f32_16x16x32_bf16(false, a0.v, false, b1.v, (short)0, acc[0][1], false, false);
        acc[1][0] = __builtin_amdgcn_wmma_f32_16x16x32_bf16(false, a1.v, false, b0.v, (short)0, acc[1][0], false, false);
        acc[1][1] = __builtin_amdgcn_wmma_f32_16x16x32_bf16(false, a1.v, false, b1.v, (short)0, acc[1][1], false, false);
    }

    // C/D layout: lane holds column n=lane&15; VGPR j holds row m = j + 8*(lane>=16)
    const int cn = lane & 15;
    const int cm = (lane >> 4) * 8;
#pragma unroll
    for (int s = 0; s < 2; ++s) {
        long n = n0 + s * 16 + cn;
        float bv = bias ? bias[n] : 0.0f;
#pragma unroll
        for (int mt = 0; mt < 2; ++mt) {
#pragma unroll
            for (int j = 0; j < 8; ++j) {
                float val = acc[mt][s][j] + bv;
                long m = m0 + mt * 16 + cm + j;
                if (C)   C[m * ldc + n] = val;
                if (Cbf) Cbf[m * ldcb + n] = (__bf16)val;
            }
        }
    }
}

// ---------------- fused GRU gate nonlinearity + state update -------------------
// gates order (torch): r, z, n.  h updated in place; bf16 mirror maintained;
// optional extra (seq) outputs for enc_out / concat buffers.
__global__ void gru_update(const float* __restrict__ gi, const float* __restrict__ gh,
                           float* h, __bf16* hb,
                           float* seq, long seq_stride,
                           __bf16* seqb, long seqb_stride) {
    int idx = blockIdx.x * blockDim.x + threadIdx.x;   // < B*H
    int b = idx >> 10;            // H == 1024
    int c = idx & (kH - 1);
    long g = (long)b * kH3 + c;
    float r = 1.f / (1.f + __expf(-(gi[g] + gh[g])));
    float z = 1.f / (1.f + __expf(-(gi[g + kH] + gh[g + kH])));
    float n = tanhf(gi[g + 2 * kH] + r * gh[g + 2 * kH]);
    float out = (1.f - z) * n + z * h[idx];
    h[idx] = out;
    hb[idx] = (__bf16)out;
    if (seq)  seq[(long)b * seq_stride + c] = out;
    if (seqb) seqb[(long)b * seqb_stride + c] = (__bf16)out;
}

// ---------------- attention score: score[b,s] = <proj[b,s,:], d1[b,:]> ---------
__global__ __launch_bounds__(32)
void attn_score(const float* __restrict__ proj, const float* __restrict__ d1,
                const int* __restrict__ src_ids, float* __restrict__ score) {
    int bs = blockIdx.x;          // b*S + s
    int b = bs >> 6;              // S == 64
    int lane = threadIdx.x;
    const float* p  = proj + (long)bs * kH;
    const float* hv = d1 + (long)b * kH;
    float acc = 0.f;
    for (int i = lane; i < kH; i += 32) acc += p[i] * hv[i];
    for (int off = 16; off; off >>= 1) acc += __shfl_down(acc, off, 32);
    if (lane == 0) score[bs] = (src_ids[bs] != 0) ? acc : -1e9f;
}

// ---------------- softmax over S=64 (in place) ----------------
__global__ __launch_bounds__(64)
void softmax64(float* score) {
    __shared__ float red[64];
    int b = blockIdx.x, t = threadIdx.x;
    float v = score[b * 64 + t];
    red[t] = v; __syncthreads();
    for (int off = 32; off; off >>= 1) { if (t < off) red[t] = fmaxf(red[t], red[t + off]); __syncthreads(); }
    float m = red[0]; __syncthreads();
    float e = __expf(v - m);
    red[t] = e; __syncthreads();
    for (int off = 32; off; off >>= 1) { if (t < off) red[t] += red[t + off]; __syncthreads(); }
    score[b * 64 + t] = e / red[0];
}

// ---------------- context: ctx[b,h] = sum_s attn[b,s]*enc_out[b,s,h] -----------
// writes bf16 context into decoder-input concat and logits concat buffers
__global__ void ctx_kernel(const float* __restrict__ attn, const float* __restrict__ enc_out,
                           __bf16* __restrict__ xdec_ctx, long xstride,
                           __bf16* __restrict__ cat_ctx, long cstride) {
    int h = blockIdx.x * blockDim.x + threadIdx.x;
    int b = blockIdx.y;
    const float* a  = attn + (long)b * kS;
    const float* eo = enc_out + (long)b * kS * kH + h;
    float acc = 0.f;
#pragma unroll 4
    for (int s = 0; s < kS; ++s) acc += a[s] * eo[(long)s * kH];
    xdec_ctx[(long)b * xstride + h] = (__bf16)acc;
    cat_ctx[(long)b * cstride + h]  = (__bf16)acc;
}

// =========================== host launcher =====================================
extern "C" void kernel_launch(void* const* d_in, const int* in_sizes, int n_in,
                              void* d_out, int out_size, void* d_ws, size_t ws_size,
                              hipStream_t stream) {
    const int*   src_ids  = (const int*)d_in[0];
    const int*   tgt_ids  = (const int*)d_in[2];
    const float* enc_emb  = (const float*)d_in[3];
    const float* dec_emb  = (const float*)d_in[4];
    const float* enc_wih0 = (const float*)d_in[5];
    const float* enc_whh0 = (const float*)d_in[6];
    const float* enc_bih0 = (const float*)d_in[7];
    const float* enc_bhh0 = (const float*)d_in[8];
    const float* enc_wih1 = (const float*)d_in[9];
    const float* enc_whh1 = (const float*)d_in[10];
    const float* enc_bih1 = (const float*)d_in[11];
    const float* enc_bhh1 = (const float*)d_in[12];
    const float* dec_wih0 = (const float*)d_in[13];
    const float* dec_whh0 = (const float*)d_in[14];
    const float* dec_bih0 = (const float*)d_in[15];
    const float* dec_bhh0 = (const float*)d_in[16];
    const float* dec_wih1 = (const float*)d_in[17];
    const float* dec_whh1 = (const float*)d_in[18];
    const float* dec_bih1 = (const float*)d_in[19];
    const float* dec_bhh1 = (const float*)d_in[20];
    const float* attn_w   = (const float*)d_in[21];
    const float* out_w    = (const float*)d_in[22];
    const float* out_b    = (const float*)d_in[23];
    float* out = (float*)d_out;

    // ---- carve workspace (256B aligned slices) ----
    char* ws = (char*)d_ws;
    size_t off = 0;
    auto carve = [&](size_t bytes) -> char* {
        char* p = ws + off;
        off = (off + bytes + 255) & ~(size_t)255;
        return p;
    };
    const size_t HB = sizeof(__bf16);
    __bf16* w_ewih0 = (__bf16*)carve(HB * (size_t)kH3 * kE);
    __bf16* w_ewhh0 = (__bf16*)carve(HB * (size_t)kH3 * kH);
    __bf16* w_ewih1 = (__bf16*)carve(HB * (size_t)kH3 * kH);
    __bf16* w_ewhh1 = (__bf16*)carve(HB * (size_t)kH3 * kH);
    __bf16* w_dwih0 = (__bf16*)carve(HB * (size_t)kH3 * (kE + kH));
    __bf16* w_dwhh0 = (__bf16*)carve(HB * (size_t)kH3 * kH);
    __bf16* w_dwih1 = (__bf16*)carve(HB * (size_t)kH3 * kH);
    __bf16* w_dwhh1 = (__bf16*)carve(HB * (size_t)kH3 * kH);
    __bf16* w_attn  = (__bf16*)carve(HB * (size_t)kH * kH);
    __bf16* w_out   = (__bf16*)carve(HB * (size_t)kV * 2 * kH);   // 131 MB: L2-resident
    __bf16* x_enc   = (__bf16*)carve(HB * (size_t)kB * kS * kE);
    float*  enc_f   = (float*)carve(sizeof(float) * (size_t)kB * kS * kH);
    __bf16* enc_bf  = (__bf16*)carve(HB * (size_t)kB * kS * kH);
    float*  proj_f  = (float*)carve(sizeof(float) * (size_t)kB * kS * kH);
    float*  gi      = (float*)carve(sizeof(float) * (size_t)kB * kH3);
    float*  gh      = (float*)carve(sizeof(float) * (size_t)kB * kH3);
    float*  h0      = (float*)carve(sizeof(float) * (size_t)kB * kH);
    float*  h1      = (float*)carve(sizeof(float) * (size_t)kB * kH);
    __bf16* h0b     = (__bf16*)carve(HB * (size_t)kB * kH);
    __bf16* h1b     = (__bf16*)carve(HB * (size_t)kB * kH);
    __bf16* xdec    = (__bf16*)carve(HB * (size_t)kB * (kE + kH));
    __bf16* catb    = (__bf16*)carve(HB * (size_t)kB * 2 * kH);
    float*  attnb   = (float*)carve(sizeof(float) * (size_t)kB * kS);
    (void)ws_size; (void)in_sizes; (void)n_in; (void)out_size;

    // ---- weights -> bf16 (done every call: deterministic, no cached state) ----
    auto cvt = [&](const float* s, __bf16* d, size_t n) {
        cvt_bf16<<<2048, 256, 0, stream>>>(s, d, (long)n);
    };
    cvt(enc_wih0, w_ewih0, (size_t)kH3 * kE);
    cvt(enc_whh0, w_ewhh0, (size_t)kH3 * kH);
    cvt(enc_wih1, w_ewih1, (size_t)kH3 * kH);
    cvt(enc_whh1, w_ewhh1, (size_t)kH3 * kH);
    cvt(dec_wih0, w_dwih0, (size_t)kH3 * (kE + kH));
    cvt(dec_whh0, w_dwhh0, (size_t)kH3 * kH);
    cvt(dec_wih1, w_dwih1, (size_t)kH3 * kH);
    cvt(dec_whh1, w_dwhh1, (size_t)kH3 * kH);
    cvt(attn_w,  w_attn,  (size_t)kH * kH);
    cvt(out_w,   w_out,   (size_t)kV * 2 * kH);

    zero_state<<<kB * kH / 256, 256, 0, stream>>>(h0, h1, h0b, h1b);

    // ---- encoder embedding gather: (B*S, E) bf16 ----
    embed_gather<<<dim3(kE / 256, kB * kS), 256, 0, stream>>>(src_ids, 1, enc_emb, x_enc, kE, kE);

    const dim3 gGate(kH3 / 32, kB / 32);   // 96 x 1 waves of 32x32 tiles
    // ---- encoder: 64 steps x 2 GRU layers ----
    for (int t = 0; t < kS; ++t) {
        // layer 0: gi = x_t @ wih0^T + bih0 ; gh = h0 @ whh0^T + bhh0
        wmma_gemm_bf16<<<gGate, 32, 0, stream>>>(x_enc + (size_t)t * kE, (long)kS * kE,
                                                 w_ewih0, kE, enc_bih0, gi, kH3, nullptr, 0, kE);
        wmma_gemm_bf16<<<gGate, 32, 0, stream>>>(h0b, kH, w_ewhh0, kH, enc_bhh0, gh, kH3, nullptr, 0, kH);
        gru_update<<<kB * kH / 256, 256, 0, stream>>>(gi, gh, h0, h0b, nullptr, 0, nullptr, 0);
        // layer 1 (h1 also streamed into enc_out fp32 + bf16)
        wmma_gemm_bf16<<<gGate, 32, 0, stream>>>(h0b, kH, w_ewih1, kH, enc_bih1, gi, kH3, nullptr, 0, kH);
        wmma_gemm_bf16<<<gGate, 32, 0, stream>>>(h1b, kH, w_ewhh1, kH, enc_bhh1, gh, kH3, nullptr, 0, kH);
        gru_update<<<kB * kH / 256, 256, 0, stream>>>(gi, gh, h1, h1b,
                                                      enc_f + (size_t)t * kH, (long)kS * kH,
                                                      enc_bf + (size_t)t * kH, (long)kS * kH);
    }

    // ---- attention projection, batched once: proj = enc_out @ attn_w^T  (M=2048) ----
    wmma_gemm_bf16<<<dim3(kH / 32, kB * kS / 32), 32, 0, stream>>>(enc_bf, kH, w_attn, kH, nullptr,
                                                                   proj_f, kH, nullptr, 0, kH);

    // ---- decoder: 47 steps (teacher forcing) ----
    for (int t = 0; t < kT - 1; ++t) {
        embed_gather<<<dim3(kE / 256, kB), 256, 0, stream>>>(tgt_ids + t, kT, dec_emb, xdec, kE + kH, kE);
        attn_score<<<kB * kS, 32, 0, stream>>>(proj_f, h1, src_ids, attnb);
        softmax64<<<kB, 64, 0, stream>>>(attnb);
        ctx_kernel<<<dim3(kH / 256, kB), 256, 0, stream>>>(attnb, enc_f,
                                                           xdec + kE, kE + kH,   // decoder input concat
                                                           catb + kH, 2 * kH);   // logits concat
        // decoder layer 0 (input = [emb, ctx], K = E+H)
        wmma_gemm_bf16<<<gGate, 32, 0, stream>>>(xdec, kE + kH, w_dwih0, kE + kH, dec_bih0, gi, kH3, nullptr, 0, kE + kH);
        wmma_gemm_bf16<<<gGate, 32, 0, stream>>>(h0b, kH, w_dwhh0, kH, dec_bhh0, gh, kH3, nullptr, 0, kH);
        gru_update<<<kB * kH / 256, 256, 0, stream>>>(gi, gh, h0, h0b, nullptr, 0, nullptr, 0);
        // decoder layer 1 (new d1 also written into catb[:, 0:H] as bf16)
        wmma_gemm_bf16<<<gGate, 32, 0, stream>>>(h0b, kH, w_dwih1, kH, dec_bih1, gi, kH3, nullptr, 0, kH);
        wmma_gemm_bf16<<<gGate, 32, 0, stream>>>(h1b, kH, w_dwhh1, kH, dec_bhh1, gh, kH3, nullptr, 0, kH);
        gru_update<<<kB * kH / 256, 256, 0, stream>>>(gi, gh, h1, h1b, nullptr, 0, catb, 2 * kH);
        // logits: (B, 2H) @ (2H, V) + out_b -> d_out[:, t, :]  (ldc = (T-1)*V)
        wmma_gemm_bf16<<<dim3(kV / 32, 1), 32, 0, stream>>>(catb, 2 * kH, w_out, 2 * kH, out_b,
                                                            out + (size_t)t * kV, (long)(kT - 1) * kV,
                                                            nullptr, 0, 2 * kH);
    }
}
